// GroupQueryAttention_84026740179395
// MI455X (gfx1250) — matrix operand order repaired and checked by reading, not compiled, not run on hardware
//
#include <hip/hip_runtime.h>
#include <hip/hip_bf16.h>

typedef __bf16 bf16_t;
typedef __attribute__((ext_vector_type(16))) __bf16 v16bf;
typedef __attribute__((ext_vector_type(8)))  __bf16 v8bf;
typedef __attribute__((ext_vector_type(4)))  __bf16 v4bf;
typedef __attribute__((ext_vector_type(8)))  float  v8f;
typedef __attribute__((ext_vector_type(4)))  float  v4f;

#define D_MODEL_      1024
#define NUM_KV_HEADS_ 4
#define GROUP_        4
#define D_K_          64
#define D_KV_         256
#define BATCH_        2
#define SEQ_          2048
#define MTOT_         (BATCH_ * SEQ_)   // 4096 rows total

static __device__ __forceinline__ bf16_t f2b(float f) { return (bf16_t)f; }

static __device__ __forceinline__ v16bf join8(v8bf lo, v8bf hi) {
  v16bf r;
#pragma unroll
  for (int j = 0; j < 8; ++j) { r[j] = lo[j]; r[8 + j] = hi[j]; }
  return r;
}

// -----------------------------------------------------------------------------
// Elementwise f32 -> bf16 (one pass; keeps all GEMM hot loops conversion-free).
// -----------------------------------------------------------------------------
__global__ __launch_bounds__(256) void cvt_f32_bf16_kernel(
    const float* __restrict__ src, bf16_t* __restrict__ dst, int n4) {
  const int i = blockIdx.x * 256 + threadIdx.x;
  if (i >= n4) return;
  v4f x = *(const v4f*)(src + (size_t)i * 4);
  v4bf y;
#pragma unroll
  for (int j = 0; j < 4; ++j) y[j] = f2b(x[j]);
  *(v4bf*)(dst + (size_t)i * 4) = y;
}

// -----------------------------------------------------------------------------
// GEMM: Out[M,N] = (A[M,K] @ W[N,K]^T + bias) * scale     (A, W already bf16)
// One wave computes a 32(M) x 64(N) tile; K-loop in steps of 32 (bf16 WMMA).
// A fragment: lane m=L%16; j<8 -> k=(L/16)*8+j, j>=8 -> 16+(L/16)*8+j-8.
// B fragment: lane n=L%16; j<8 -> k=(L/16)*8+j, j>=8 -> 16+(L/16)*8+j-8 (as A).
// C fragment: lane n=L%16; vgpr r -> m = r + 8*(L/16).
// -----------------------------------------------------------------------------
template <bool OUT_BF16>
__global__ __launch_bounds__(32) void gemm_bias_wmma(
    const bf16_t* __restrict__ A, const bf16_t* __restrict__ W,
    const float* __restrict__ bias, void* __restrict__ Outptr,
    int N, int K, float scale) {
  const int lane = threadIdx.x & 31;
  const int hf = lane >> 4;
  const int lr = lane & 15;
  const int m0 = blockIdx.y * 32;
  const int n0 = blockIdx.x * 64;

  const bf16_t* arow0 = A + (size_t)(m0 + lr) * K;
  const bf16_t* arow1 = arow0 + (size_t)16 * K;

  v8f acc[2][4] = {};

  for (int k0 = 0; k0 < K; k0 += 32) {
    v16bf af0 = join8(*(const v8bf*)(arow0 + k0 + hf * 8),
                      *(const v8bf*)(arow0 + k0 + hf * 8 + 16));
    v16bf af1 = join8(*(const v8bf*)(arow1 + k0 + hf * 8),
                      *(const v8bf*)(arow1 + k0 + hf * 8 + 16));
#pragma unroll
    for (int t = 0; t < 4; ++t) {
      const bf16_t* w = W + (size_t)(n0 + t * 16 + lr) * K + k0 + hf * 8;
      v16bf wf = join8(*(const v8bf*)w, *(const v8bf*)(w + 16));
      acc[0][t] = __builtin_amdgcn_wmma_f32_16x16x32_bf16(
          false, af0, false, wf, (short)0, acc[0][t], false, false);
      acc[1][t] = __builtin_amdgcn_wmma_f32_16x16x32_bf16(
          false, af1, false, wf, (short)0, acc[1][t], false, false);
    }
  }

#pragma unroll
  for (int t = 0; t < 4; ++t) {
    const int n = n0 + t * 16 + lr;
    const float bval = bias[n];
#pragma unroll
    for (int mt = 0; mt < 2; ++mt) {
#pragma unroll
      for (int r = 0; r < 8; ++r) {
        const size_t m = (size_t)m0 + mt * 16 + hf * 8 + r;
        const float v = (acc[mt][t][r] + bval) * scale;
        if (OUT_BF16) ((bf16_t*)Outptr)[m * N + n] = f2b(v);
        else          ((float*)Outptr)[m * N + n] = v;
      }
    }
  }
}

// -----------------------------------------------------------------------------
// V transpose: Vp[B*S, 256] -> Vt[B, Hkv, 64, S] so the P@V B-operand
// (indexed n=d, k=key) loads contiguous key runs.
// -----------------------------------------------------------------------------
__global__ __launch_bounds__(256) void transpose_v_kernel(
    const bf16_t* __restrict__ Vp, bf16_t* __restrict__ Vt) {
  const int idx = blockIdx.x * 256 + threadIdx.x;    // over B*S*256 = 2^20
  if (idx >= MTOT_ * D_KV_) return;
  const int c = idx & (D_KV_ - 1);
  const int s = (idx >> 8) & (SEQ_ - 1);
  const int b = idx >> 19;
  const int h = c >> 6;
  const int d = c & 63;
  Vt[(((size_t)(b * NUM_KV_HEADS_ + h)) * D_K_ + d) * SEQ_ + s] = Vp[idx];
}

// -----------------------------------------------------------------------------
// Flash attention, one wave per (b, kv-head h, group g, 16 query rows).
// Scores computed TRANSPOSED: St = K_tile(A) x Q_tile(B), so the C layout puts
// query = lane (n) and key = vgpr (m). Row softmax stats become in-register
// reductions + ONE shfl_xor(16) to merge the two half-waves, and the exp'd
// probabilities land directly in the A-fragment layout needed for P @ V
// (no LDS transpose). Only alpha / l are broadcast through 16 floats of LDS.
// Qp is pre-scaled by 1/sqrt(D_K).
// -----------------------------------------------------------------------------
__global__ __launch_bounds__(32) void gqa_flash_kernel(
    const bf16_t* __restrict__ Qp,   // [B*S, 1024] bf16, pre-scaled
    const bf16_t* __restrict__ Kp,   // [B*S, 256]  bf16
    const bf16_t* __restrict__ Vt,   // [B, 4, 64, S] bf16
    bf16_t* __restrict__ Ao) {       // [B*S, 1024] bf16
  __shared__ __align__(16) float alds[16];
  const int lane = threadIdx.x & 31;
  const int hf = lane >> 4;
  const int lr = lane & 15;

  int bid = blockIdx.x;
  const int mt = bid & (SEQ_ / 16 - 1); bid >>= 7;
  const int g  = bid & 3;               bid >>= 2;
  const int h  = bid & 3;               bid >>= 2;
  const int b  = bid;
  const int qcol = (h * GROUP_ + g) * D_K_;
  const int row0 = b * SEQ_ + mt * 16;

  // Q as B-operand: lane holds query n=lr, k = d = step*32 + hf*8 + j (j<8), step*32 + 16 + hf*8 + j-8 (j>=8).
  v16bf qf0, qf1;
  {
    const bf16_t* qrow = Qp + (size_t)(row0 + lr) * D_MODEL_ + qcol;
    qf0 = join8(*(const v8bf*)(qrow + hf * 8),
                *(const v8bf*)(qrow + hf * 8 + 16));
    qf1 = join8(*(const v8bf*)(qrow + 32 + hf * 8),
                *(const v8bf*)(qrow + 32 + hf * 8 + 16));
  }

  float mrow = -3.0e38f;   // running max for query lr (dup across halves)
  float lrow = 0.0f;       // running denom for query lr
  v8f oacc[4] = {};        // O tile [16 q x 64 d]: lane = d col, vgpr = q row

  const bf16_t* vhead = Vt + (size_t)(b * NUM_KV_HEADS_ + h) * D_K_ * SEQ_;

  for (int t0 = 0; t0 < SEQ_; t0 += 32) {
    // K as A-operand: lane key m = lr (+16 for tile b), k = d.
    const bf16_t* ka = Kp + (size_t)(b * SEQ_ + t0 + lr) * D_KV_ + h * D_K_;
    const bf16_t* kb = ka + 16 * D_KV_;
    if (t0 + 32 < SEQ_) {
      __builtin_prefetch(ka + 32 * D_KV_, 0, 0);
      __builtin_prefetch(kb + 32 * D_KV_, 0, 0);
    }

    v8f sta = {}, stb = {};
    sta = __builtin_amdgcn_wmma_f32_16x16x32_bf16(false,
        join8(*(const v8bf*)(ka + hf * 8), *(const v8bf*)(ka + hf * 8 + 16)),
        false, qf0, (short)0, sta, false, false);
    sta = __builtin_amdgcn_wmma_f32_16x16x32_bf16(false,
        join8(*(const v8bf*)(ka + 32 + hf * 8), *(const v8bf*)(ka + 32 + hf * 8 + 16)),
        false, qf1, (short)0, sta, false, false);
    stb = __builtin_amdgcn_wmma_f32_16x16x32_bf16(false,
        join8(*(const v8bf*)(kb + hf * 8), *(const v8bf*)(kb + hf * 8 + 16)),
        false, qf0, (short)0, stb, false, false);
    stb = __builtin_amdgcn_wmma_f32_16x16x32_bf16(false,
        join8(*(const v8bf*)(kb + 32 + hf * 8), *(const v8bf*)(kb + 32 + hf * 8 + 16)),
        false, qf1, (short)0, stb, false, false);

    // Per-lane row max over this lane's 16 keys, then merge halves (1 shfl).
    float vmax = fmaxf(sta[0], stb[0]);
#pragma unroll
    for (int j = 1; j < 8; ++j) vmax = fmaxf(vmax, fmaxf(sta[j], stb[j]));
    vmax = fmaxf(vmax, __shfl_xor(vmax, 16, 32));

    const float mn = fmaxf(mrow, vmax);
    const float alpha = __expf(mrow - mn);
    if (lane < 16) alds[lr] = alpha;   // broadcast alpha[q] for O rescale

    // P values: tile-a vgpr j == A-frag k=hf*8+j ; tile-b == k=16+hf*8+j.
    float pA[8], pB[8], psum = 0.0f;
#pragma unroll
    for (int j = 0; j < 8; ++j) {
      pA[j] = __expf(sta[j] - mn);
      pB[j] = __expf(stb[j] - mn);
      psum += pA[j] + pB[j];
    }
    psum += __shfl_xor(psum, 16, 32);
    lrow = lrow * alpha + psum;
    mrow = mn;

    v16bf pf;
#pragma unroll
    for (int j = 0; j < 8; ++j) { pf[j] = f2b(pA[j]); pf[8 + j] = f2b(pB[j]); }

    // Rescale O accumulator rows by alpha (rows = queries hf*8+r).
    asm volatile("s_wait_dscnt 0" ::: "memory");
    v4f al0 = *(const v4f*)(alds + hf * 8);
    v4f al1 = *(const v4f*)(alds + hf * 8 + 4);
#pragma unroll
    for (int nd = 0; nd < 4; ++nd) {
#pragma unroll
      for (int r = 0; r < 4; ++r) {
        oacc[nd][r]     *= al0[r];
        oacc[nd][4 + r] *= al1[r];
      }
    }

    // O += P[16x32] @ V[32x64] over 4 d-subtiles.
#pragma unroll
    for (int nd = 0; nd < 4; ++nd) {
      const bf16_t* vrow = vhead + (size_t)(nd * 16 + lr) * SEQ_ + t0 + hf * 8;
      v16bf vf = join8(*(const v8bf*)vrow, *(const v8bf*)(vrow + 16));
      oacc[nd] = __builtin_amdgcn_wmma_f32_16x16x32_bf16(
          false, pf, false, vf, (short)0, oacc[nd], false, false);
    }
    asm volatile("" ::: "memory");  // keep next iter's LDS write after reads
  }

  // Broadcast l[q] and normalize.
  if (lane < 16) alds[lr] = lrow;
  asm volatile("s_wait_dscnt 0" ::: "memory");
  v4f l0 = *(const v4f*)(alds + hf * 8);
  v4f l1 = *(const v4f*)(alds + hf * 8 + 4);
  float linv[8];
#pragma unroll
  for (int r = 0; r < 4; ++r) { linv[r] = 1.0f / l0[r]; linv[4 + r] = 1.0f / l1[r]; }

#pragma unroll
  for (int nd = 0; nd < 4; ++nd) {
#pragma unroll
    for (int r = 0; r < 8; ++r) {
      const size_t m = (size_t)row0 + hf * 8 + r;
      Ao[m * D_MODEL_ + qcol + nd * 16 + lr] = f2b(oacc[nd][r] * linv[r]);
    }
  }
}

extern "C" void kernel_launch(void* const* d_in, const int* in_sizes, int n_in,
                              void* d_out, int out_size, void* d_ws,
                              size_t ws_size, hipStream_t stream) {
  (void)in_sizes; (void)n_in; (void)out_size; (void)ws_size;

  const float* query = (const float*)d_in[0];
  const float* key   = (const float*)d_in[1];
  const float* value = (const float*)d_in[2];
  const float* w_q   = (const float*)d_in[3];
  const float* b_q   = (const float*)d_in[4];
  const float* w_k   = (const float*)d_in[5];
  const float* b_k   = (const float*)d_in[6];
  const float* w_v   = (const float*)d_in[7];
  const float* b_v   = (const float*)d_in[8];
  const float* w_o   = (const float*)d_in[9];
  const float* b_o   = (const float*)d_in[10];
  float* out = (float*)d_out;

  // Workspace layout (bf16 elements), ~43 MB total.
  bf16_t* p   = (bf16_t*)d_ws;
  bf16_t* qx  = p; p += (size_t)MTOT_ * D_MODEL_;   // bf16 query
  bf16_t* kx  = p; p += (size_t)MTOT_ * D_MODEL_;   // bf16 key
  bf16_t* vx  = p; p += (size_t)MTOT_ * D_MODEL_;   // bf16 value
  bf16_t* wqb = p; p += (size_t)D_MODEL_ * D_MODEL_;
  bf16_t* wkb = p; p += (size_t)D_KV_ * D_MODEL_;
  bf16_t* wvb = p; p += (size_t)D_KV_ * D_MODEL_;
  bf16_t* wob = p; p += (size_t)D_MODEL_ * D_MODEL_;
  bf16_t* Qp  = p; p += (size_t)MTOT_ * D_MODEL_;
  bf16_t* Kp  = p; p += (size_t)MTOT_ * D_KV_;
  bf16_t* Vp  = p; p += (size_t)MTOT_ * D_KV_;
  bf16_t* Vt  = p; p += (size_t)MTOT_ * D_KV_;
  bf16_t* Ao  = qx;   // qx is dead once Qp is built; reuse for attention out

  const dim3 wave(32);
  const float qscale = 0.125f;  // 1/sqrt(64) folded into the Q projection

  // One-shot f32 -> bf16 conversions.
  const int nact4 = (MTOT_ * D_MODEL_) / 4;          // 1M vec4
  const int nwq4  = (D_MODEL_ * D_MODEL_) / 4;       // 256K vec4
  const int nwk4  = (D_KV_ * D_MODEL_) / 4;          // 64K vec4
  cvt_f32_bf16_kernel<<<nact4 / 256, 256, 0, stream>>>(query, qx, nact4);
  cvt_f32_bf16_kernel<<<nact4 / 256, 256, 0, stream>>>(key,   kx, nact4);
  cvt_f32_bf16_kernel<<<nact4 / 256, 256, 0, stream>>>(value, vx, nact4);
  cvt_f32_bf16_kernel<<<nwq4 / 256, 256, 0, stream>>>(w_q, wqb, nwq4);
  cvt_f32_bf16_kernel<<<nwk4 / 256, 256, 0, stream>>>(w_k, wkb, nwk4);
  cvt_f32_bf16_kernel<<<nwk4 / 256, 256, 0, stream>>>(w_v, wvb, nwk4);
  cvt_f32_bf16_kernel<<<nwq4 / 256, 256, 0, stream>>>(w_o, wob, nwq4);

  // Projections (bf16 x bf16 -> bf16).
  gemm_bias_wmma<true><<<dim3(D_MODEL_ / 64, MTOT_ / 32), wave, 0, stream>>>(
      qx, wqb, b_q, Qp, D_MODEL_, D_MODEL_, qscale);
  gemm_bias_wmma<true><<<dim3(D_KV_ / 64, MTOT_ / 32), wave, 0, stream>>>(
      kx, wkb, b_k, Kp, D_KV_, D_MODEL_, 1.0f);
  gemm_bias_wmma<true><<<dim3(D_KV_ / 64, MTOT_ / 32), wave, 0, stream>>>(
      vx, wvb, b_v, Vp, D_KV_, D_MODEL_, 1.0f);

  // Per-head V transpose.
  transpose_v_kernel<<<(MTOT_ * D_KV_) / 256, 256, 0, stream>>>(Vp, Vt);

  // Flash attention: 2 * 4 * 4 * 128 = 4096 waves.
  gqa_flash_kernel<<<BATCH_ * NUM_KV_HEADS_ * GROUP_ * (SEQ_ / 16), wave, 0,
                     stream>>>(Qp, Kp, Vt, Ao);

  // Output projection (bf16 in, f32 out) straight into d_out.
  gemm_bias_wmma<false><<<dim3(D_MODEL_ / 64, MTOT_ / 32), wave, 0, stream>>>(
      Ao, wob, b_o, out, D_MODEL_, D_MODEL_, 1.0f);
}
